// MEFKTSequenceModel_12421045420885
// MI455X (gfx1250) — matrix operand order repaired and checked by reading, not compiled, not run on hardware
//
#include <hip/hip_runtime.h>
#include <hip/hip_bf16.h>
#include <math.h>

// ---------------------------------------------------------------------------
// Types for CDNA5 WMMA (wave32): D(16x16 f32) = A(16x32 bf16) x B(32x16 bf16) + C
// ---------------------------------------------------------------------------
typedef __attribute__((ext_vector_type(16))) __bf16 bfrag;   // 8 VGPRs
typedef __attribute__((ext_vector_type(8)))  float  v8f;     // 8 VGPRs
typedef __attribute__((ext_vector_type(4)))  unsigned int uv4;
typedef unsigned short u16;

union Frag { bfrag f; uv4 q[2]; };

__device__ __forceinline__ v8f wmma_bf16(bfrag a, bfrag b, v8f c) {
  // (neg_a, A, neg_b, B, c_mod, C, reuse_a, reuse_b)
  return __builtin_amdgcn_wmma_f32_16x16x32_bf16(false, a, false, b, (short)0, c,
                                                 false, false);
}

__device__ __forceinline__ u16 f2bf(float f) {
  unsigned int u = __float_as_uint(f);
  unsigned int r = (u + 0x7FFFu + ((u >> 16) & 1u)) >> 16;
  return (u16)r;
}
__device__ __forceinline__ float bf2f(u16 h) {
  return __uint_as_float(((unsigned int)h) << 16);
}

// Generic wave GEMM accumulate: A is a 16 x ldA bf16 tile (row-major, LDS),
// Brow points at W^T row n (bf16, [n][Kd] contiguous in K).
// Lane layout: rsel = lane&15, ksel = lane>>4.
__device__ __forceinline__ v8f wgemm(const u16* A, int ldA, int Kd,
                                     const u16* Brow, int rsel, int ksel, v8f acc) {
  for (int k0 = 0; k0 < Kd; k0 += 32) {
    Frag a, b;
    const u16* ap = A + rsel * ldA + k0 + ksel * 16;  // A: lane row, 16 contiguous K
    a.q[0] = *(const uv4*)ap;
    a.q[1] = *(const uv4*)(ap + 8);
    const u16* bp = Brow + k0 + ksel * 16;            // B: lane col n, 16 contiguous K
    b.q[0] = *(const uv4*)bp;
    b.q[1] = *(const uv4*)(bp + 8);
    acc = wmma_bf16(a.f, b.f, acc);
  }
  return acc;
}

// ---------------------------------------------------------------------------
// Weight transpose + convert: W [K][N] f32 -> Wt [N][K] bf16
// ---------------------------------------------------------------------------
__global__ void transpose_bf16(const float* __restrict__ W, u16* __restrict__ Wt,
                               int K, int N) {
  int gid = blockIdx.x * blockDim.x + threadIdx.x;
  if (gid >= K * N) return;
  int k = gid / N, n = gid % N;
  Wt[(size_t)n * K + k] = f2bf(W[gid]);
}

// ---------------------------------------------------------------------------
// Embedding gather + Q/K/inter/V GEMMs. 16-row tiles, 8 waves, 16-col N tiles.
// ---------------------------------------------------------------------------
__global__ void __launch_bounds__(256)
embed_gemm(const int* __restrict__ idx, const int* __restrict__ flg,
           const float* __restrict__ item_emb, const float* __restrict__ ans_emb,
           const u16* __restrict__ WqT, const float* __restrict__ bq,
           const u16* __restrict__ WkT, const float* __restrict__ bk,
           const u16* __restrict__ WvT, const float* __restrict__ bv,
           const u16* __restrict__ WhT, const float* __restrict__ bh,
           u16* __restrict__ ebf, u16* __restrict__ Qb, u16* __restrict__ Kb,
           u16* __restrict__ Vt) {
  __shared__ __attribute__((aligned(16))) u16 eA[16][128];
  __shared__ __attribute__((aligned(16))) u16 aA[16][128];
  __shared__ __attribute__((aligned(16))) u16 iA[16][128];
  int tid = threadIdx.x;
  int r0 = blockIdx.x * 16;  // flattened (b,s) row base; S=1024 divisible by 16

  for (int k = 0; k < 8; k++) {
    int e = tid + k * 256;
    int row = e >> 7, col = e & 127;
    int it = idx[r0 + row];
    u16 ev = f2bf(item_emb[(size_t)it * 128 + col]);
    eA[row][col] = ev;
    ebf[(size_t)(r0 + row) * 128 + col] = ev;
    int fl = flg[r0 + row];
    aA[row][col] = f2bf(ans_emb[(size_t)fl * 128 + col]);
  }
  __syncthreads();

  int wv = tid >> 5, lane = tid & 31, rsel = lane & 15, ksel = lane >> 4;
  int mhalf = ksel * 8;
  int n = wv * 16 + rsel;
  v8f zero8 = {0.f, 0.f, 0.f, 0.f, 0.f, 0.f, 0.f, 0.f};

  // Q = e @ Wq + bq
  {
    v8f acc = wgemm(&eA[0][0], 128, 128, WqT + (size_t)n * 128, rsel, ksel, zero8);
    float bn = bq[n];
#pragma unroll
    for (int j = 0; j < 8; j++)
      Qb[(size_t)(r0 + j + mhalf) * 128 + n] = f2bf(acc[j] + bn);
  }
  // K = e @ Wk + bk
  {
    v8f acc = wgemm(&eA[0][0], 128, 128, WkT + (size_t)n * 128, rsel, ksel, zero8);
    float bn = bk[n];
#pragma unroll
    for (int j = 0; j < 8; j++)
      Kb[(size_t)(r0 + j + mhalf) * 128 + n] = f2bf(acc[j] + bn);
  }
  // inter = [e|a] @ Wh + bh  (K = 256 split into two 128 halves)
  {
    v8f acc = wgemm(&eA[0][0], 128, 128, WhT + (size_t)n * 256, rsel, ksel, zero8);
    acc = wgemm(&aA[0][0], 128, 128, WhT + (size_t)n * 256 + 128, rsel, ksel, acc);
    float bn = bh[n];
#pragma unroll
    for (int j = 0; j < 8; j++)
      iA[j + mhalf][n] = f2bf(acc[j] + bn);
  }
  __syncthreads();
  // V = inter @ Wv + bv -> stored transposed Vt[b][h][d][s]
  {
    v8f acc = wgemm(&iA[0][0], 128, 128, WvT + (size_t)n * 128, rsel, ksel, zero8);
    float bn = bv[n];
    int h = n >> 5, d = n & 31;
#pragma unroll
    for (int j = 0; j < 8; j++) {
      int row = r0 + j + mhalf;
      int b = row >> 10, s = row & 1023;
      Vt[((size_t)((b * 4 + h) * 32 + d)) * 1024 + s] = f2bf(acc[j] + bn);
    }
  }
}

// ---------------------------------------------------------------------------
// sr[b,h,s] = scale * dot(Q[b,s,h,:], K[b,s,h,:])
// ---------------------------------------------------------------------------
__global__ void sr_kernel(const u16* __restrict__ Qb, const u16* __restrict__ Kb,
                          float* __restrict__ sr) {
  int gid = blockIdx.x * 256 + threadIdx.x;  // 16384 = 4096 rows * 4 heads
  int row = gid >> 2, h = gid & 3;
  const u16* q = Qb + (size_t)row * 128 + h * 32;
  const u16* k = Kb + (size_t)row * 128 + h * 32;
  float s = 0.f;
#pragma unroll 8
  for (int d = 0; d < 32; d++) s += bf2f(q[d]) * bf2f(k[d]);
  int b = row >> 10, sidx = row & 1023;
  sr[(size_t)(b * 4 + h) * 1024 + sidx] = s * 0.17677669529663687f;
}

// Ep[bh][0..1024]: Ep[0]=0, Ep[p]=sum exp(sr - max) over s<p  (serial, tiny)
__global__ void scan_ep(const float* __restrict__ sr, float* __restrict__ Ep) {
  if (threadIdx.x != 0) return;
  int bh = blockIdx.x;
  const float* s = sr + (size_t)bh * 1024;
  float mx = -3.0e38f;
  for (int i = 0; i < 1024; i++) mx = fmaxf(mx, s[i]);
  float* e = Ep + (size_t)bh * 1025;
  float run = 0.f;
  e[0] = 0.f;
  for (int i = 0; i < 1024; i++) { run += expf(s[i] - mx); e[i + 1] = run; }
}

// Pp[b][0..1024]: prefix sums of max(gap, 1)
__global__ void scan_pp(const float* __restrict__ g, float* __restrict__ Pp) {
  if (threadIdx.x != 0) return;
  int b = blockIdx.x;
  const float* gb = g + (size_t)b * 1024;
  float* p = Pp + (size_t)b * 1025;
  float run = 0.f;
  p[0] = 0.f;
  for (int i = 0; i < 1024; i++) { run += fmaxf(gb[i], 1.f); p[i + 1] = run; }
}

// ---------------------------------------------------------------------------
// Attention with decay. One wave per (b, h, 16-row t tile). Two passes:
// pass1 row-max (WMMA scores + decay, shfl_xor reduce), pass2 exp + p@V WMMA.
// ---------------------------------------------------------------------------
__global__ void __launch_bounds__(256)
attn_kernel(const u16* __restrict__ Qb, const u16* __restrict__ Kb,
            const u16* __restrict__ Vt, const float* __restrict__ Pp,
            const float* __restrict__ Ep, const float* __restrict__ theta_raw,
            u16* __restrict__ ctxb) {
  __shared__ __attribute__((aligned(16))) u16 pb[8][16][32];
  int tid = threadIdx.x, wv = tid >> 5, lane = tid & 31;
  int rsel = lane & 15, ksel = lane >> 4, mhalf = ksel * 8;
  int blk = blockIdx.x;            // 128 blocks
  int b = blk >> 5, rem = blk & 31;
  int h = rem >> 3, tg = rem & 7;
  int tile = tg * 8 + wv;          // 0..63
  int tp0 = tile * 16 + 1;         // actual sequence position of first row
  float theta = log1pf(expf(theta_raw[h])) + 1e-4f;
  const float scale = 0.17677669529663687f;
  const float* Ppb = Pp + (size_t)b * 1025;
  const float* Epb = Ep + (size_t)(b * 4 + h) * 1025;

  // Q A-fragment (16 rows t x 32 cols d)
  Frag qa;
  {
    int tp = min(tp0 + rsel, 1023);
    const u16* p = Qb + ((size_t)(b * 1024 + tp)) * 128 + h * 32 + ksel * 16;
    qa.q[0] = *(const uv4*)p;
    qa.q[1] = *(const uv4*)(p + 8);
  }
  int tpi[8];
  float Ppt[8], Einv[8];
#pragma unroll
  for (int j = 0; j < 8; j++) {
    int tp = tp0 + j + mhalf;
    int tpc = min(tp, 1023);
    tpi[j] = tp;
    Ppt[j] = Ppb[tpc];
    Einv[j] = 1.f / Epb[tpc];
  }
  int ilim = min(1024, (tile + 1) * 16);  // i < tp <= 16*(tile+1)
  v8f zero8 = {0.f, 0.f, 0.f, 0.f, 0.f, 0.f, 0.f, 0.f};

  // ---- Pass 1: row max ----
  float mx[8];
#pragma unroll
  for (int j = 0; j < 8; j++) mx[j] = -3.0e38f;
  for (int i0 = 0; i0 < ilim; i0 += 16) {
    Frag kb;
    {
      const u16* p = Kb + ((size_t)(b * 1024 + i0 + rsel)) * 128 + h * 32 + ksel * 16;
      kb.q[0] = *(const uv4*)p;
      kb.q[1] = *(const uv4*)(p + 8);
    }
    v8f sv = wmma_bf16(qa.f, kb.f, zero8);
    int in_ = i0 + rsel;
    float Ppi = Ppb[in_], Epi = Epb[in_];
#pragma unroll
    for (int j = 0; j < 8; j++) {
      float pd = (float)(tpi[j] - in_) * (Ppt[j] - Ppi) * (1.f - Epi * Einv[j]);
      float val = sv[j] * scale * expf(-theta * pd);
      val = (in_ < tpi[j]) ? val : -1e30f;
      mx[j] = fmaxf(mx[j], val);
    }
  }
#pragma unroll
  for (int j = 0; j < 8; j++) {
    mx[j] = fmaxf(mx[j], __shfl_xor(mx[j], 1, 32));
    mx[j] = fmaxf(mx[j], __shfl_xor(mx[j], 2, 32));
    mx[j] = fmaxf(mx[j], __shfl_xor(mx[j], 4, 32));
    mx[j] = fmaxf(mx[j], __shfl_xor(mx[j], 8, 32));
  }

  // ---- Pass 2: exp, p-transpose through LDS, p @ V ----
  v8f c0 = zero8, c1 = zero8;
  float rs[8];
#pragma unroll
  for (int j = 0; j < 8; j++) rs[j] = 0.f;

  for (int i0 = 0; i0 < ilim; i0 += 32) {
#pragma unroll
    for (int half = 0; half < 2; half++) {
      int ib = i0 + half * 16;
      Frag kb;
      {
        const u16* p = Kb + ((size_t)(b * 1024 + ib + rsel)) * 128 + h * 32 + ksel * 16;
        kb.q[0] = *(const uv4*)p;
        kb.q[1] = *(const uv4*)(p + 8);
      }
      v8f sv = wmma_bf16(qa.f, kb.f, zero8);
      int in_ = ib + rsel;
      float Ppi = Ppb[in_], Epi = Epb[in_];
#pragma unroll
      for (int j = 0; j < 8; j++) {
        float pd = (float)(tpi[j] - in_) * (Ppt[j] - Ppi) * (1.f - Epi * Einv[j]);
        float val = sv[j] * scale * expf(-theta * pd);
        val = (in_ < tpi[j]) ? val : -1e30f;
        float pe = expf(val - mx[j]);   // masked -> 0
        rs[j] += pe;
        pb[wv][j + mhalf][half * 16 + rsel] = f2bf(pe);
      }
    }
    // Ensure this wave's LDS stores are visible before the relayout loads.
    asm volatile("s_wait_dscnt 0" ::: "memory");
    Frag pa;
    {
      const u16* lp = &pb[wv][rsel][ksel * 16];
      pa.q[0] = *(const uv4*)lp;
      pa.q[1] = *(const uv4*)(lp + 8);
    }
    {
      const u16* p0 = Vt + ((size_t)((b * 4 + h) * 32 + rsel)) * 1024 + i0 + ksel * 16;
      Frag vb;
      vb.q[0] = *(const uv4*)p0;
      vb.q[1] = *(const uv4*)(p0 + 8);
      c0 = wmma_bf16(pa.f, vb.f, c0);
      const u16* p1 = Vt + ((size_t)((b * 4 + h) * 32 + 16 + rsel)) * 1024 + i0 + ksel * 16;
      vb.q[0] = *(const uv4*)p1;
      vb.q[1] = *(const uv4*)(p1 + 8);
      c1 = wmma_bf16(pa.f, vb.f, c1);
    }
  }
#pragma unroll
  for (int j = 0; j < 8; j++) {
    rs[j] += __shfl_xor(rs[j], 1, 32);
    rs[j] += __shfl_xor(rs[j], 2, 32);
    rs[j] += __shfl_xor(rs[j], 4, 32);
    rs[j] += __shfl_xor(rs[j], 8, 32);
  }
#pragma unroll
  for (int j = 0; j < 8; j++) {
    int tt = tile * 16 + j + mhalf;   // output row t in [0,1023)
    if (tt < 1023) {
      float inv = 1.f / rs[j];
      size_t base = ((size_t)(b * 1023 + tt)) * 128 + h * 32;
      ctxb[base + rsel] = f2bf(c0[j] * inv);
      ctxb[base + 16 + rsel] = f2bf(c1[j] * inv);
    }
  }
}

// ---------------------------------------------------------------------------
// Head: feat=[ctx|e_next] (256) @ W1 + b1, ReLU, @ W2 + b2, sigmoid.
// ---------------------------------------------------------------------------
__global__ void __launch_bounds__(256)
head_kernel(const u16* __restrict__ ctxb, const u16* __restrict__ ebf,
            const u16* __restrict__ W1T, const float* __restrict__ b1,
            const float* __restrict__ W2, const float* __restrict__ b2,
            float* __restrict__ out) {
  __shared__ __attribute__((aligned(16))) u16 fA[16][256];
  __shared__ float lg[16];
  const int R = 4 * 1023;
  int tid = threadIdx.x;
  int r0 = blockIdx.x * 16;
  if (tid < 16) lg[tid] = 0.f;
  for (int k = 0; k < 16; k++) {
    int e = tid + k * 256;
    int row = e >> 8, col = e & 255;
    int gr = min(r0 + row, R - 1);
    int b = gr / 1023, t = gr % 1023;
    u16 v = (col < 128) ? ctxb[((size_t)(b * 1023 + t)) * 128 + col]
                        : ebf[((size_t)(b * 1024 + t + 1)) * 128 + (col - 128)];
    fA[row][col] = v;
  }
  __syncthreads();
  int wv = tid >> 5, lane = tid & 31, rsel = lane & 15, ksel = lane >> 4;
  int mhalf = ksel * 8;
  int n = wv * 16 + rsel;
  v8f zero8 = {0.f, 0.f, 0.f, 0.f, 0.f, 0.f, 0.f, 0.f};
  v8f acc = wgemm(&fA[0][0], 256, 256, W1T + (size_t)n * 256, rsel, ksel, zero8);
  float b1n = b1[n], w2n = W2[n];
  float pj[8];
#pragma unroll
  for (int j = 0; j < 8; j++) {
    float hv = fmaxf(acc[j] + b1n, 0.f);
    pj[j] = hv * w2n;
  }
#pragma unroll
  for (int j = 0; j < 8; j++) {
    pj[j] += __shfl_xor(pj[j], 1, 32);
    pj[j] += __shfl_xor(pj[j], 2, 32);
    pj[j] += __shfl_xor(pj[j], 4, 32);
    pj[j] += __shfl_xor(pj[j], 8, 32);
  }
  if (rsel == 0) {
#pragma unroll
    for (int j = 0; j < 8; j++) atomicAdd(&lg[j + mhalf], pj[j]);
  }
  __syncthreads();
  if (tid < 16) {
    int gr = r0 + tid;
    if (gr < R) {
      float l = lg[tid] + b2[0];
      out[gr] = 1.f / (1.f + expf(-l));
      out[R + gr] = 1.f;  // valid
    }
  }
}

// ---------------------------------------------------------------------------
// Host launcher
// ---------------------------------------------------------------------------
static inline size_t alignup(size_t x) { return (x + 255) & ~(size_t)255; }

extern "C" void kernel_launch(void* const* d_in, const int* in_sizes, int n_in,
                              void* d_out, int out_size, void* d_ws, size_t ws_size,
                              hipStream_t stream) {
  const int*   idx      = (const int*)d_in[0];
  const int*   flg      = (const int*)d_in[1];
  const float* gaps     = (const float*)d_in[2];
  const float* item_emb = (const float*)d_in[3];
  const float* ans_emb  = (const float*)d_in[4];
  const float* Wq = (const float*)d_in[5];
  const float* bq = (const float*)d_in[6];
  const float* Wk = (const float*)d_in[7];
  const float* bk = (const float*)d_in[8];
  const float* Wv = (const float*)d_in[9];
  const float* bv = (const float*)d_in[10];
  const float* Wh = (const float*)d_in[11];
  const float* bh = (const float*)d_in[12];
  const float* W1 = (const float*)d_in[13];
  const float* b1 = (const float*)d_in[14];
  const float* W2 = (const float*)d_in[15];
  const float* b2 = (const float*)d_in[16];
  const float* theta_raw = (const float*)d_in[17];

  char* w = (char*)d_ws;
  u16* WqT = (u16*)w;  w += alignup(128 * 128 * 2);
  u16* WkT = (u16*)w;  w += alignup(128 * 128 * 2);
  u16* WvT = (u16*)w;  w += alignup(128 * 128 * 2);
  u16* WhT = (u16*)w;  w += alignup(256 * 128 * 2);
  u16* W1T = (u16*)w;  w += alignup(256 * 128 * 2);
  u16* ebf = (u16*)w;  w += alignup((size_t)4096 * 128 * 2);
  u16* Qb  = (u16*)w;  w += alignup((size_t)4096 * 128 * 2);
  u16* Kb  = (u16*)w;  w += alignup((size_t)4096 * 128 * 2);
  u16* Vt  = (u16*)w;  w += alignup((size_t)16 * 32 * 1024 * 2);
  u16* ctxb = (u16*)w; w += alignup((size_t)4 * 1023 * 128 * 2);
  float* sr = (float*)w; w += alignup((size_t)16 * 1024 * 4);
  float* Ep = (float*)w; w += alignup((size_t)16 * 1025 * 4);
  float* Pp = (float*)w; w += alignup((size_t)4 * 1025 * 4);

  transpose_bf16<<<64, 256, 0, stream>>>(Wq, WqT, 128, 128);
  transpose_bf16<<<64, 256, 0, stream>>>(Wk, WkT, 128, 128);
  transpose_bf16<<<64, 256, 0, stream>>>(Wv, WvT, 128, 128);
  transpose_bf16<<<128, 256, 0, stream>>>(Wh, WhT, 256, 128);
  transpose_bf16<<<128, 256, 0, stream>>>(W1, W1T, 256, 128);

  embed_gemm<<<256, 256, 0, stream>>>(idx, flg, item_emb, ans_emb,
                                      WqT, bq, WkT, bk, WvT, bv, WhT, bh,
                                      ebf, Qb, Kb, Vt);

  sr_kernel<<<64, 256, 0, stream>>>(Qb, Kb, sr);
  scan_ep<<<16, 32, 0, stream>>>(sr, Ep);
  scan_pp<<<4, 32, 0, stream>>>(gaps, Pp);

  attn_kernel<<<128, 256, 0, stream>>>(Qb, Kb, Vt, Pp, Ep, theta_raw, ctxb);

  head_kernel<<<256, 256, 0, stream>>>(ctxb, ebf, W1T, b1, W2, b2, (float*)d_out);
}
